// TransformerRoPE_90726889160945
// MI455X (gfx1250) — matrix operand order repaired and checked by reading, not compiled, not run on hardware
//
#include <hip/hip_runtime.h>
#include <hip/hip_bf16.h>
#include <stdint.h>

typedef __attribute__((ext_vector_type(16))) _Float16 v16h;
typedef __attribute__((ext_vector_type(8)))  float    v8f;
typedef __attribute__((ext_vector_type(4)))  unsigned u32x4;
typedef __attribute__((ext_vector_type(8)))  int      i32x8;
typedef __attribute__((ext_vector_type(4)))  int      i32x4;
typedef _Float16 f16;

#define L_SEQ 2048
#define NTOK  4096

#if __has_builtin(__builtin_amdgcn_tensor_load_to_lds)
#define HAVE_TDM 1
#else
#define HAVE_TDM 0
#endif

// ---------------------------------------------------------------------------
// WMMA helpers (CDNA5 gfx1250, wave32).  D = A(16x32 f16) * B(32x16 f16) + C
// ---------------------------------------------------------------------------
static __device__ __forceinline__ v8f wmma_f16(v16h a, v16h b, v8f c) {
  return __builtin_amdgcn_wmma_f32_16x16x32_f16(
      /*neg_a=*/false, a, /*neg_b=*/false, b,
      /*c_mod=*/(short)0, c, /*reuse_a=*/false, /*reuse_b=*/false);
}

// A fragment: 16(M) x 32(K), f16, LDS row-major (lda elems).  Each VGPR's two
// f16 are contiguous (even k) -> single aligned b32 load per VGPR.
static __device__ __forceinline__ v16h load_A_frag(const f16* p, int lda) {
  const int lane = threadIdx.x & 31;
  const int row  = lane & 15;
  const int half = lane >> 4;
  union { v16h v; unsigned w[8]; } f;
#pragma unroll
  for (int j = 0; j < 8; ++j) {
    const int k = (half << 4) + (j << 1);
    f.w[j] = *(const unsigned*)&p[row * lda + k];
  }
  return f.v;
}

// B fragment: 32(K) x 16(N), f16, from LDS [K][ldb] row-major; p = &lds[0][n0].
static __device__ __forceinline__ v16h load_B_frag(const f16* p, int ldb) {
  const int lane = threadIdx.x & 31;
  const int col  = lane & 15;
  const int half = lane >> 4;
  union { v16h v; f16 e[16]; } f;
#pragma unroll
  for (int j = 0; j < 8; ++j) {
    const int k = (half << 4) + (j << 1);
    f.e[2 * j]     = p[k * ldb + col];
    f.e[2 * j + 1] = p[(k + 1) * ldb + col];
  }
  return f.v;
}

// Transposed-B fragment: logical B[k][n] = lds[n][k]; p = &lds[n0][koff].
// Pairs contiguous along k -> b32 loads.
static __device__ __forceinline__ v16h load_Bt_frag(const f16* p, int ld) {
  const int lane = threadIdx.x & 31;
  const int col  = lane & 15;
  const int half = lane >> 4;
  union { v16h v; unsigned w[8]; } f;
#pragma unroll
  for (int j = 0; j < 8; ++j) {
    const int k = (half << 4) + (j << 1);
    f.w[j] = *(const unsigned*)&p[col * ld + k];
  }
  return f.v;
}

// ---------------------------------------------------------------------------
// TDM: DMA a 2D f16 tile [tile_h x tile_w] from a strided tensor into LDS.
// Builds the D# per CDNA5 ISA ch.8 (group0/group1); issued once per wave.
// ---------------------------------------------------------------------------
#if HAVE_TDM
static __device__ __forceinline__ void tdm_load_tile_f16(
    unsigned lds_addr, const void* gptr, unsigned tile_w, unsigned tile_h,
    unsigned stride_elems) {
  const unsigned long long ga = (unsigned long long)gptr;
  const unsigned td0 = stride_elems;   // tensor dim0 (row length, elems)
  const unsigned td1 = tile_h;         // tensor dim1 (#rows)
  u32x4 g0;
  g0[0] = 1u;                                   // count=1, user descriptor
  g0[1] = lds_addr;                             // LDS byte address
  g0[2] = (unsigned)(ga & 0xffffffffu);         // global_addr[31:0]
  g0[3] = (unsigned)((ga >> 32) & 0x01ffffffu)  // global_addr[56:32]
          | 0x80000000u;                        // type=2 ("image")
  i32x8 g1;
  g1[0] = (int)(1u << 16);                               // data_size=2 bytes
  g1[1] = (int)((td0 & 0xffffu) << 16);                  // tensor_dim0 lo
  g1[2] = (int)((td0 >> 16) | ((td1 & 0xffffu) << 16));  // td0 hi | td1 lo
  g1[3] = (int)((td1 >> 16) | (tile_w << 16));           // td1 hi | tile_dim0
  g1[4] = (int)(tile_h & 0xffffu);                       // tile_dim1 (dim2=0)
  g1[5] = (int)stride_elems;                             // dim0_stride[31:0]
  g1[6] = 0;
  g1[7] = 0;
  i32x4 gz = {0, 0, 0, 0};
#if __clang_major__ >= 23
  i32x8 gz8 = {0, 0, 0, 0, 0, 0, 0, 0};
  __builtin_amdgcn_tensor_load_to_lds(g0, g1, gz, gz, gz8, 0);
#else
  __builtin_amdgcn_tensor_load_to_lds(g0, g1, gz, gz, 0);
#endif
}
#endif

static __device__ __forceinline__ unsigned lds_addr_of(const void* p) {
  return (unsigned)(unsigned long long)(uintptr_t)p;  // low 32b = LDS offset
}

// Async global->LDS copy, 16B per lane (VGLOBAL, tracked by ASYNCcnt).
static __device__ __forceinline__ void async_copy_b128(unsigned lds_dst,
                                                       const void* gsrc) {
  const unsigned long long ga = (unsigned long long)gsrc;
  asm volatile("global_load_async_to_lds_b128 %0, %1, off"
               :: "v"(lds_dst), "v"(ga) : "memory");
}
static __device__ __forceinline__ void async_wait0() {
  asm volatile("s_wait_asynccnt 0x0" ::: "memory");
}
// <=8 outstanding: current double-buffer stage complete (in-order completion)
static __device__ __forceinline__ void async_wait8() {
  asm volatile("s_wait_asynccnt 0x8" ::: "memory");
}

// ---------------------------------------------------------------------------
// Elementwise / prep kernels
// ---------------------------------------------------------------------------
__global__ void k_f32_to_f16(const float* __restrict__ in, f16* __restrict__ out,
                             size_t n) {
  size_t i = (size_t)blockIdx.x * blockDim.x + threadIdx.x;
  if (i < n) out[i] = (f16)in[i];
}

// LayerNorm over rows of [NTOK, H], f32 in -> f16 out.  One block per row.
__global__ __launch_bounds__(256) void k_ln(const float* __restrict__ x,
                                            const float* __restrict__ g,
                                            const float* __restrict__ b,
                                            f16* __restrict__ out, int H) {
  __shared__ float s1[8], s2[8];
  const int row = blockIdx.x;
  const float* xr = x + (size_t)row * H;
  float sum = 0.f, sq = 0.f;
  for (int i = threadIdx.x; i < H; i += 256) {
    float v = xr[i];
    sum += v; sq += v * v;
  }
#pragma unroll
  for (int off = 16; off > 0; off >>= 1) {
    sum += __shfl_xor(sum, off, 32);
    sq  += __shfl_xor(sq,  off, 32);
  }
  if ((threadIdx.x & 31) == 0) { s1[threadIdx.x >> 5] = sum; s2[threadIdx.x >> 5] = sq; }
  __syncthreads();
  float ts = 0.f, tq = 0.f;
#pragma unroll
  for (int i = 0; i < 8; ++i) { ts += s1[i]; tq += s2[i]; }
  const float mu   = ts / (float)H;
  const float var  = tq / (float)H - mu * mu;
  const float rinv = rsqrtf(var + 1e-5f);
  for (int i = threadIdx.x; i < H; i += 256)
    out[(size_t)row * H + i] = (f16)((xr[i] - mu) * rinv * g[i] + b[i]);
}

// RoPE + head split.  kqv f16 [NTOK, 3072] (chunk order k,q,v) ->
// q/k/v f16 [B*NH][L][64].  One thread per (token, head, pair).
__global__ void k_rope(const f16* __restrict__ kqv, f16* __restrict__ qo,
                       f16* __restrict__ ko, f16* __restrict__ vo, int total) {
  int idx = blockIdx.x * blockDim.x + threadIdx.x;
  if (idx >= total) return;
  const int p = idx & 31;          // pair 0..31
  const int h = (idx >> 5) & 15;   // head
  const int t = idx >> 9;          // token 0..4095
  const int b = t / L_SEQ, l = t % L_SEQ;
  const float inv = __expf(-(float)p * 0.28782314f);  // ln(10000)/32
  const float ang = (float)l * inv;
  float s, c;
  __sincosf(ang, &s, &c);
  const size_t base = (size_t)t * 3072 + h * 64 + 2 * p;
  __builtin_prefetch(&kqv[base + 3072], 0, 1);
  const float k0 = (float)kqv[base],        k1 = (float)kqv[base + 1];
  const float q0 = (float)kqv[base + 1024], q1 = (float)kqv[base + 1025];
  const float v0 = (float)kqv[base + 2048], v1 = (float)kqv[base + 2049];
  const size_t o = ((size_t)(b * 16 + h) * L_SEQ + l) * 64 + 2 * p;
  qo[o] = (f16)(q0 * c - q1 * s); qo[o + 1] = (f16)(q1 * c + q0 * s);
  ko[o] = (f16)(k0 * c - k1 * s); ko[o + 1] = (f16)(k1 * c + k0 * s);
  vo[o] = (f16)v0;                vo[o + 1] = (f16)v1;
}

// ---------------------------------------------------------------------------
// Generic WMMA GEMM with double-buffered TDM staging.
// wave0 issues A-tile DMAs, wave1 issues B-tile DMAs for stage t+1 while all
// waves run WMMA on stage t; s_wait_tensorcnt(1) = current tile landed.
// ---------------------------------------------------------------------------
__global__ __launch_bounds__(256) void k_gemm_f16(
    const f16* __restrict__ A, int lda, const f16* __restrict__ B, int ldb,
    int K, float* __restrict__ Cf, f16* __restrict__ Ch, int ldc,
    const float* __restrict__ addm, const float* __restrict__ bias) {
  __shared__ __align__(16) f16 As[2][64 * 32];
  __shared__ __align__(16) f16 Bs[2][32 * 64];
  const int tid  = threadIdx.x;
  const int wave = tid >> 5;
  const int lane = tid & 31;
  const int n0 = blockIdx.x * 64;
  const int m0 = blockIdx.y * 64;
  const int wm = (wave >> 1) << 4;  // 0,16,32,48
  const int wn = (wave & 1) << 5;   // 0,32
  v8f acc0 = {}, acc1 = {};
#if HAVE_TDM
  if (wave == 0) tdm_load_tile_f16(lds_addr_of(As[0]), &A[m0 * lda], 32, 64, lda);
  if (wave == 1) tdm_load_tile_f16(lds_addr_of(Bs[0]), &B[n0], 64, 32, ldb);
#endif
  for (int k0 = 0; k0 < K; k0 += 32) {
    const int cur = (k0 >> 5) & 1;
    __syncthreads();  // previous stage's consumers done; buf[1-cur] is free
#if HAVE_TDM
    const bool more = (k0 + 32) < K;
    if (more) {
      if (wave == 0)
        tdm_load_tile_f16(lds_addr_of(As[1 - cur]), &A[m0 * lda + k0 + 32],
                          32, 64, lda);
      if (wave == 1)
        tdm_load_tile_f16(lds_addr_of(Bs[1 - cur]), &B[(k0 + 32) * ldb + n0],
                          64, 32, ldb);
    }
    if (wave < 2) {
      if (more) __builtin_amdgcn_s_wait_tensorcnt(1);
      else      __builtin_amdgcn_s_wait_tensorcnt(0);
    }
#else
    {
      const int ar = tid >> 2, ac = (tid & 3) << 3;
      const int br = tid >> 3, bc = (tid & 7) << 3;
      *(uint4*)&As[cur][ar * 32 + ac] =
          *(const uint4*)&A[(m0 + ar) * lda + k0 + ac];
      *(uint4*)&Bs[cur][br * 64 + bc] =
          *(const uint4*)&B[(k0 + br) * ldb + n0 + bc];
    }
#endif
    __syncthreads();  // publish current stage
    v16h af = load_A_frag(&As[cur][wm * 32], 32);
    v16h b0 = load_B_frag(&Bs[cur][wn], 64);
    v16h b1 = load_B_frag(&Bs[cur][wn + 16], 64);
    acc0 = wmma_f16(af, b0, acc0);
    acc1 = wmma_f16(af, b1, acc1);
  }
  const int col = lane & 15;
  const int rh  = (lane >> 4) << 3;  // 0 or 8
#pragma unroll
  for (int t = 0; t < 2; ++t) {
    v8f acc = t ? acc1 : acc0;
    const int cc = n0 + wn + t * 16 + col;
#pragma unroll
    for (int r = 0; r < 8; ++r) {
      const int rr = m0 + wm + rh + r;
      float v = acc[r];
      if (bias) v += bias[cc];
      const size_t idx = (size_t)rr * ldc + cc;
      if (addm) v += addm[idx];
      if (Cf) Cf[idx] = v; else Ch[idx] = (f16)v;
    }
  }
}

// ---------------------------------------------------------------------------
// Fused FFN1 + SwiGLU: out = (A@B[:,n]+b[n]) * swish(A@B[:,n+halfN]+b[n+halfN])
// Double-buffered TDM staging; waves 0/1/2 issue A / B_a / B_gate tiles.
// ---------------------------------------------------------------------------
__global__ __launch_bounds__(256) void k_ffn1_swiglu(
    const f16* __restrict__ A, int lda, const f16* __restrict__ B, int ldb,
    const float* __restrict__ bias, f16* __restrict__ out, int ldo,
    int K, int halfN) {
  __shared__ __align__(16) f16 As[2][64 * 32];
  __shared__ __align__(16) f16 Ba[2][32 * 64];
  __shared__ __align__(16) f16 Bg[2][32 * 64];
  const int tid  = threadIdx.x;
  const int wave = tid >> 5;
  const int lane = tid & 31;
  const int n0 = blockIdx.x * 64;
  const int m0 = blockIdx.y * 64;
  const int wm = (wave >> 1) << 4;
  const int wn = (wave & 1) << 5;
  v8f aa0 = {}, aa1 = {}, ag0 = {}, ag1 = {};
#if HAVE_TDM
  if (wave == 0) tdm_load_tile_f16(lds_addr_of(As[0]), &A[m0 * lda], 32, 64, lda);
  if (wave == 1) tdm_load_tile_f16(lds_addr_of(Ba[0]), &B[n0], 64, 32, ldb);
  if (wave == 2) tdm_load_tile_f16(lds_addr_of(Bg[0]), &B[halfN + n0], 64, 32, ldb);
#endif
  for (int k0 = 0; k0 < K; k0 += 32) {
    const int cur = (k0 >> 5) & 1;
    __syncthreads();
#if HAVE_TDM
    const bool more = (k0 + 32) < K;
    if (more) {
      if (wave == 0)
        tdm_load_tile_f16(lds_addr_of(As[1 - cur]), &A[m0 * lda + k0 + 32],
                          32, 64, lda);
      if (wave == 1)
        tdm_load_tile_f16(lds_addr_of(Ba[1 - cur]), &B[(k0 + 32) * ldb + n0],
                          64, 32, ldb);
      if (wave == 2)
        tdm_load_tile_f16(lds_addr_of(Bg[1 - cur]),
                          &B[(k0 + 32) * ldb + halfN + n0], 64, 32, ldb);
    }
    if (wave < 3) {
      if (more) __builtin_amdgcn_s_wait_tensorcnt(1);
      else      __builtin_amdgcn_s_wait_tensorcnt(0);
    }
#else
    {
      const int ar = tid >> 2, ac = (tid & 3) << 3;
      const int br = tid >> 3, bc = (tid & 7) << 3;
      *(uint4*)&As[cur][ar * 32 + ac] =
          *(const uint4*)&A[(m0 + ar) * lda + k0 + ac];
      *(uint4*)&Ba[cur][br * 64 + bc] =
          *(const uint4*)&B[(k0 + br) * ldb + n0 + bc];
      *(uint4*)&Bg[cur][br * 64 + bc] =
          *(const uint4*)&B[(k0 + br) * ldb + halfN + n0 + bc];
    }
#endif
    __syncthreads();
    v16h af  = load_A_frag(&As[cur][wm * 32], 32);
    v16h ba0 = load_B_frag(&Ba[cur][wn], 64);
    v16h ba1 = load_B_frag(&Ba[cur][wn + 16], 64);
    v16h bg0 = load_B_frag(&Bg[cur][wn], 64);
    v16h bg1 = load_B_frag(&Bg[cur][wn + 16], 64);
    aa0 = wmma_f16(af, ba0, aa0);
    aa1 = wmma_f16(af, ba1, aa1);
    ag0 = wmma_f16(af, bg0, ag0);
    ag1 = wmma_f16(af, bg1, ag1);
  }
  const int col = lane & 15;
  const int rh  = (lane >> 4) << 3;
#pragma unroll
  for (int t = 0; t < 2; ++t) {
    v8f accA = t ? aa1 : aa0;
    v8f accG = t ? ag1 : ag0;
    const int cc = n0 + wn + t * 16 + col;
#pragma unroll
    for (int r = 0; r < 8; ++r) {
      const int rr = m0 + wm + rh + r;
      float a = accA[r] + bias[cc];
      float g = accG[r] + bias[halfN + cc];
      float sw = a * g * (1.f / (1.f + __expf(-g)));
      out[(size_t)rr * ldo + cc] = (f16)sw;
    }
  }
}

// ---------------------------------------------------------------------------
// Flash attention.  Grid: (L/64 query blocks, B*NH heads).  128 thr = 4 waves.
// K/V tiles double-buffered via async global->LDS copies: each wave issues the
// next block's 8 b128 copies, then s_wait_asynccnt 8 (in-order completion
// guarantees the current block's copies retired).
// ---------------------------------------------------------------------------
__global__ __launch_bounds__(128) void k_attn(
    const f16* __restrict__ Q, const f16* __restrict__ Kk,
    const f16* __restrict__ V, f16* __restrict__ ctx, int L) {
  __shared__ __align__(16) f16 Qs[64 * 64];
  __shared__ __align__(16) f16 Ks[2][64 * 64];
  __shared__ __align__(16) f16 Vs[2][64 * 64];
  __shared__ __align__(16) f16 Ps[4 * 16 * 64];
  const int tid  = threadIdx.x;
  const int wave = tid >> 5;
  const int lane = tid & 31;
  const int bh = blockIdx.y;
  const int m0 = blockIdx.x * 64;
  const f16* Qh = Q  + (size_t)bh * L * 64;
  const f16* Kh = Kk + (size_t)bh * L * 64;
  const f16* Vh = V  + (size_t)bh * L * 64;
  // stage Q tile [64,64] asynchronously
  for (int i = tid; i < 512; i += 128) {
    int r = i >> 3, c = (i & 7) << 3;
    async_copy_b128(lds_addr_of(&Qs[r * 64 + c]), &Qh[(size_t)(m0 + r) * 64 + c]);
  }
  async_wait0();
  // prologue: stage K/V block 0 into buffer 0
  for (int i = tid; i < 512; i += 128) {
    int r = i >> 3, c = (i & 7) << 3;
    async_copy_b128(lds_addr_of(&Ks[0][r * 64 + c]), &Kh[(size_t)r * 64 + c]);
    async_copy_b128(lds_addr_of(&Vs[0][r * 64 + c]), &Vh[(size_t)r * 64 + c]);
  }
  const int wr  = wave << 4;
  const int col = lane & 15;
  const int rh  = (lane >> 4) << 3;
  float rowm[8], rowl[8];
#pragma unroll
  for (int r = 0; r < 8; ++r) { rowm[r] = -1e30f; rowl[r] = 0.f; }
  v8f O[4] = {};
  for (int kb = 0; kb < L; kb += 64) {
    const int cur = (kb >> 6) & 1;
    __syncthreads();  // all waves done with buf[1-cur]
    const bool more = (kb + 64) < L;
    if (more) {
      const int nb = kb + 64;
      for (int i = tid; i < 512; i += 128) {
        int r = i >> 3, c = (i & 7) << 3;
        async_copy_b128(lds_addr_of(&Ks[1 - cur][r * 64 + c]),
                        &Kh[(size_t)(nb + r) * 64 + c]);
        async_copy_b128(lds_addr_of(&Vs[1 - cur][r * 64 + c]),
                        &Vh[(size_t)(nb + r) * 64 + c]);
      }
      async_wait8();  // current block's copies complete; next still in flight
    } else {
      async_wait0();
    }
    __syncthreads();  // publish current block
    // S = (Q K^T) * 1/sqrt(64)
    v16h aq0 = load_A_frag(&Qs[wr * 64], 64);
    v16h aq1 = load_A_frag(&Qs[wr * 64 + 32], 64);
    v8f S[4];
#pragma unroll
    for (int nt = 0; nt < 4; ++nt) {
      v16h b0 = load_Bt_frag(&Ks[cur][(nt * 16) * 64], 64);
      v16h b1 = load_Bt_frag(&Ks[cur][(nt * 16) * 64 + 32], 64);
      v8f s = {};
      s = wmma_f16(aq0, b0, s);
      s = wmma_f16(aq1, b1, s);
      S[nt] = s;
    }
    // online softmax (C-layout row = rh + r)
#pragma unroll
    for (int r = 0; r < 8; ++r) {
      float sv[4];
      float m = rowm[r];
#pragma unroll
      for (int nt = 0; nt < 4; ++nt) {
        sv[nt] = S[nt][r] * 0.125f;
        m = fmaxf(m, sv[nt]);
      }
#pragma unroll
      for (int off = 1; off < 16; off <<= 1) m = fmaxf(m, __shfl_xor(m, off, 32));
      const float alpha = __expf(rowm[r] - m);
      float psum = 0.f;
#pragma unroll
      for (int nt = 0; nt < 4; ++nt) {
        float p = __expf(sv[nt] - m);
        psum += p;
        Ps[wave * 1024 + (rh + r) * 64 + nt * 16 + col] = (f16)p;
      }
#pragma unroll
      for (int off = 1; off < 16; off <<= 1) psum += __shfl_xor(psum, off, 32);
      rowl[r] = rowl[r] * alpha + psum;
      rowm[r] = m;
#pragma unroll
      for (int nt = 0; nt < 4; ++nt) O[nt][r] *= alpha;
    }
    __syncthreads();
    // O += P @ V
    v16h ap0 = load_A_frag(&Ps[wave * 1024], 64);
    v16h ap1 = load_A_frag(&Ps[wave * 1024 + 32], 64);
#pragma unroll
    for (int nt = 0; nt < 4; ++nt) {
      v16h b0 = load_B_frag(&Vs[cur][nt * 16], 64);
      v16h b1 = load_B_frag(&Vs[cur][32 * 64 + nt * 16], 64);
      O[nt] = wmma_f16(ap0, b0, O[nt]);
      O[nt] = wmma_f16(ap1, b1, O[nt]);
    }
  }
  // write ctx: token-major [NTOK, 1024], col = head*64 + d
  const int b = bh >> 4, h = bh & 15;
#pragma unroll
  for (int nt = 0; nt < 4; ++nt)
#pragma unroll
    for (int r = 0; r < 8; ++r) {
      float v = O[nt][r] / rowl[r];
      size_t row = (size_t)(b * L + m0 + wr + rh + r);
      ctx[row * 1024 + h * 64 + nt * 16 + col] = (f16)v;
    }
}

// ---------------------------------------------------------------------------
// Host launch
// ---------------------------------------------------------------------------
extern "C" void kernel_launch(void* const* d_in, const int* in_sizes, int n_in,
                              void* d_out, int out_size, void* d_ws,
                              size_t ws_size, hipStream_t stream) {
  (void)in_sizes; (void)n_in; (void)out_size; (void)ws_size;
  const float* x      = (const float*)d_in[0];
  const float* w_kqv  = (const float*)d_in[1];
  const float* w_out  = (const float*)d_in[2];
  const float* w_ffn1 = (const float*)d_in[3];
  const float* b_ffn1 = (const float*)d_in[4];
  const float* w_ffn2 = (const float*)d_in[5];
  const float* b_ffn2 = (const float*)d_in[6];
  const float* ln1_g  = (const float*)d_in[7];
  const float* ln1_b  = (const float*)d_in[8];
  const float* ln2_g  = (const float*)d_in[9];
  const float* ln2_b  = (const float*)d_in[10];
  float* out = (float*)d_out;

  char* ws = (char*)d_ws;
  // swiglu [4096,4096] overlaps pre1 [4096,1024] + kqv [4096,3072] (dead then)
  f16* swiglu = (f16*)ws;
  f16* pre1   = (f16*)ws;
  f16* kqv    = (f16*)(ws + 8388608);
  size_t off = 33554432;
  f16*   wkqv_h  = (f16*)(ws + off); off += 6291456;
  f16*   q_r     = (f16*)(ws + off); off += 8388608;
  f16*   k_r     = (f16*)(ws + off); off += 8388608;
  f16*   v_r     = (f16*)(ws + off); off += 8388608;
  f16*   ctx     = (f16*)(ws + off); off += 8388608;
  f16*   wout_h  = (f16*)(ws + off); off += 2097152;
  float* resid   = (float*)(ws + off); off += 16777216;
  f16*   pre2    = (f16*)(ws + off); off += 8388608;
  f16*   wffn1_h = (f16*)(ws + off); off += 16777216;
  f16*   wffn2_h = (f16*)(ws + off); off += 8388608;

  // 1) weight conversions f32 -> f16
  {
    size_t n;
    n = (size_t)1024 * 3072;
    k_f32_to_f16<<<(unsigned)((n + 255) / 256), 256, 0, stream>>>(w_kqv, wkqv_h, n);
    n = (size_t)1024 * 1024;
    k_f32_to_f16<<<(unsigned)((n + 255) / 256), 256, 0, stream>>>(w_out, wout_h, n);
    n = (size_t)1024 * 8192;
    k_f32_to_f16<<<(unsigned)((n + 255) / 256), 256, 0, stream>>>(w_ffn1, wffn1_h, n);
    n = (size_t)4096 * 1024;
    k_f32_to_f16<<<(unsigned)((n + 255) / 256), 256, 0, stream>>>(w_ffn2, wffn2_h, n);
  }
  // 2) LN1 -> pre1 (f16)
  k_ln<<<NTOK, 256, 0, stream>>>(x, ln1_g, ln1_b, pre1, 1024);
  // 3) kqv = pre1 @ w_kqv  -> f16 [4096, 3072]
  k_gemm_f16<<<dim3(3072 / 64, NTOK / 64), 256, 0, stream>>>(
      pre1, 1024, wkqv_h, 3072, 1024, nullptr, kqv, 3072, nullptr, nullptr);
  // 4) RoPE + head split
  {
    int total = NTOK * 16 * 32;
    k_rope<<<(total + 255) / 256, 256, 0, stream>>>(kqv, q_r, k_r, v_r, total);
  }
  // 5) attention -> ctx f16 [4096, 1024]
  k_attn<<<dim3(L_SEQ / 64, 32), 128, 0, stream>>>(q_r, k_r, v_r, ctx, L_SEQ);
  // 6) resid = x + ctx @ w_out   (f32)
  k_gemm_f16<<<dim3(1024 / 64, NTOK / 64), 256, 0, stream>>>(
      ctx, 1024, wout_h, 1024, 1024, resid, nullptr, 1024, x, nullptr);
  // 7) LN2 -> pre2 (f16)
  k_ln<<<NTOK, 256, 0, stream>>>(resid, ln2_g, ln2_b, pre2, 1024);
  // 8) fused FFN1 + SwiGLU -> swiglu f16 [4096, 4096]
  k_ffn1_swiglu<<<dim3(4096 / 64, NTOK / 64), 256, 0, stream>>>(
      pre2, 1024, wffn1_h, 8192, b_ffn1, swiglu, 4096, 1024, 4096);
  // 9) out = resid + swiglu @ w_ffn2 + b_ffn2   (f32)
  k_gemm_f16<<<dim3(1024 / 64, NTOK / 64), 256, 0, stream>>>(
      swiglu, 4096, wffn2_h, 1024, 4096, out, nullptr, 1024, resid, b_ffn2);
}
